// Latent_40853728919856
// MI455X (gfx1250) — matrix operand order repaired and checked by reading, not compiled, not run on hardware
//
#include <hip/hip_runtime.h>
#include <hip/hip_bf16.h>

typedef __attribute__((ext_vector_type(16))) __bf16 v16bf;
typedef __attribute__((ext_vector_type(8)))  __bf16 v8bf;
typedef __attribute__((ext_vector_type(8)))  float  v8f;

#define NZ 16384
#define NE 8192
#define DIM 256
#define INF_BITS 0x7F800000u

// ---------------------------------------------------------------------------
// ws layout:
//   [0,      64KB)   z2       f32[16384]
//   [64KB,   96KB)   e2       f32[8192]
//   [96KB,  160KB)   rowmin   u32[16384]
//   [160KB, 192KB)   colmin   u32[8192]
//   [192KB, ...  )   optional bf16 split arrays: zhi, zlo (8MB each),
//                    ehi, elo (4MB each)  -> fast path if ws_size allows
// ---------------------------------------------------------------------------

__global__ __launch_bounds__(256) void vq_init_kernel(unsigned int* __restrict__ minbuf) {
    int i = blockIdx.x * 256 + threadIdx.x;
    if (i < NZ + NE) minbuf[i] = INF_BITS;
}

// one wave32 per row: squared norms
__global__ __launch_bounds__(256) void vq_norms_kernel(const float* __restrict__ z,
                                                       const float* __restrict__ e,
                                                       float* __restrict__ z2,
                                                       float* __restrict__ e2) {
    int lane = threadIdx.x & 31;
    int wid  = (blockIdx.x * 256 + threadIdx.x) >> 5;
    const float* src;
    float* dst;
    int row;
    if (wid < NZ) { src = z; dst = z2; row = wid; }
    else          { src = e; dst = e2; row = wid - NZ; }
    const float* p = src + (long)row * DIM;
    float s = 0.0f;
#pragma unroll
    for (int j = 0; j < 8; ++j) {
        float v = p[lane + 32 * j];
        s = fmaf(v, v, s);
    }
#pragma unroll
    for (int m = 1; m < 32; m <<= 1) s += __shfl_xor(s, m, 32);
    if (lane == 0) dst[row] = s;
}

// f32 -> (bf16 hi, bf16 residual) split, 8 elements per thread
__global__ __launch_bounds__(256) void vq_split_kernel(const float* __restrict__ z,
                                                       const float* __restrict__ e,
                                                       __bf16* __restrict__ zhi,
                                                       __bf16* __restrict__ zlo,
                                                       __bf16* __restrict__ ehi,
                                                       __bf16* __restrict__ elo) {
    long idx = (long)blockIdx.x * 256 + threadIdx.x;
    const long zgroups = (long)NZ * DIM / 8;
    const float* src;
    __bf16 *dhi, *dlo;
    long off;
    if (idx < zgroups) { src = z; dhi = zhi; dlo = zlo; off = idx * 8; }
    else               { src = e; dhi = ehi; dlo = elo; off = (idx - zgroups) * 8; }
    float4 u0 = *(const float4*)(src + off);
    float4 u1 = *(const float4*)(src + off + 4);
    float f[8] = {u0.x, u0.y, u0.z, u0.w, u1.x, u1.y, u1.z, u1.w};
    v8bf h, l;
#pragma unroll
    for (int i = 0; i < 8; ++i) {
        __bf16 hh = (__bf16)f[i];
        h[i] = hh;
        l[i] = (__bf16)(f[i] - (float)hh);
    }
    *(v8bf*)(dhi + off) = h;
    *(v8bf*)(dlo + off) = l;
}

// fuse two 8-half loads into one v16bf operand (pure register renaming)
__device__ __forceinline__ v16bf ld16bf(const __bf16* p0, const __bf16* p1) {
    v8bf a = *(const v8bf*)p0;
    v8bf b = *(const v8bf*)p1;
    return __builtin_shufflevector(a, b, 0, 1, 2, 3, 4, 5, 6, 7,
                                   8, 9, 10, 11, 12, 13, 14, 15);
}

// split f32 -> bf16 hi + residual in-register (fallback path)
__device__ __forceinline__ void cvt_split8(float4 u, float4 v, v16bf& hi, v16bf& lo, int base) {
    float f[8] = {u.x, u.y, u.z, u.w, v.x, v.y, v.z, v.w};
#pragma unroll
    for (int i = 0; i < 8; ++i) {
        __bf16 h = (__bf16)f[i];
        hi[base + i] = h;
        lo[base + i] = (__bf16)(f[i] - (float)h);
    }
}

// shared epilogue: d2 = max(z2 + e2 - 2*cross, 0); min-reduce both axes
__device__ __forceinline__ void vq_epilogue(v8f* acc, const float* __restrict__ z2,
                                            const float* __restrict__ e2,
                                            unsigned int* __restrict__ rowmin,
                                            unsigned int* __restrict__ colmin,
                                            unsigned int* cmin_lds,
                                            int rowbase, int colbase,
                                            int l16, int lg, int tid) {
    float z2r[8];
#pragma unroll
    for (int r = 0; r < 8; ++r) z2r[r] = z2[rowbase + r + 8 * lg];

    float pm[8];
#pragma unroll
    for (int r = 0; r < 8; ++r) pm[r] = __builtin_inff();

#pragma unroll
    for (int t = 0; t < 16; ++t) {
        float e2c = e2[colbase + t * 16 + l16];
        float cm = __builtin_inff();
#pragma unroll
        for (int r = 0; r < 8; ++r) {
            float d = fmaxf(z2r[r] + e2c - 2.0f * acc[t][r], 0.0f);
            pm[r] = fminf(pm[r], d);
            cm = fminf(cm, d);
        }
        cm = fminf(cm, __shfl_xor(cm, 16, 32));    // merge the two 8-row halves
        if (lg == 0) atomicMin(&cmin_lds[t * 16 + l16], __float_as_uint(cm));
    }

#pragma unroll
    for (int r = 0; r < 8; ++r) {
        float v = pm[r];
        v = fminf(v, __shfl_xor(v, 1, 32));
        v = fminf(v, __shfl_xor(v, 2, 32));
        v = fminf(v, __shfl_xor(v, 4, 32));
        v = fminf(v, __shfl_xor(v, 8, 32));
        if (l16 == 0)
            atomicMin(&rowmin[rowbase + r + 8 * lg], __float_as_uint(v));
    }

    __syncthreads();
    atomicMin(&colmin[colbase + tid], cmin_lds[tid]);
}

// -------- fast path: operands pre-split to bf16 hi/lo in ws; hot loop is
// pure global_load_b128 + v_wmma (no conversion VALU) ------------------------
__global__ __launch_bounds__(256) void vq_wmma_fast_kernel(const __bf16* __restrict__ zhi,
                                                           const __bf16* __restrict__ zlo,
                                                           const __bf16* __restrict__ ehi,
                                                           const __bf16* __restrict__ elo,
                                                           const float* __restrict__ z2,
                                                           const float* __restrict__ e2,
                                                           unsigned int* __restrict__ rowmin,
                                                           unsigned int* __restrict__ colmin) {
    __shared__ unsigned int cmin_lds[256];
    const int tid = threadIdx.x;
    cmin_lds[tid] = INF_BITS;
    __syncthreads();

    const int lane = tid & 31;
    const int wave = tid >> 5;
    const int l16  = lane & 15;
    const int lg   = lane >> 4;
    const int colbase = blockIdx.x * 256;
    const int rowbase = blockIdx.y * 128 + wave * 16;

    v8f acc[16];
#pragma unroll
    for (int t = 0; t < 16; ++t) acc[t] = (v8f){0.f,0.f,0.f,0.f,0.f,0.f,0.f,0.f};

    // A: lane row = rowbase+l16; K chunks {8lg..8lg+7} and {8lg+16..8lg+23}
    const __bf16* zh = zhi + (long)(rowbase + l16) * DIM + lg * 8;
    const __bf16* zl = zlo + (long)(rowbase + l16) * DIM + lg * 8;
    // B: lane col = colbase + 16t + l16; K chunks {8lg..8lg+7} and {8lg+16..8lg+23}
    const __bf16* eh0 = ehi + (long)(colbase + l16) * DIM + lg * 8;
    const __bf16* el0 = elo + (long)(colbase + l16) * DIM + lg * 8;

#pragma unroll 1
    for (int s = 0; s < DIM / 32; ++s) {
        const int k0 = s * 32;
        v16bf a_hi = ld16bf(zh + k0, zh + k0 + 16);
        v16bf a_lo = ld16bf(zl + k0, zl + k0 + 16);
#pragma unroll
        for (int t = 0; t < 16; ++t) {
            const long toff = (long)t * 16 * DIM + k0;
            v16bf b_hi = ld16bf(eh0 + toff, eh0 + toff + 16);
            v16bf b_lo = ld16bf(el0 + toff, el0 + toff + 16);
            acc[t] = __builtin_amdgcn_wmma_f32_16x16x32_bf16(false, a_hi, false, b_hi,
                                                             (short)0, acc[t], false, false);
            acc[t] = __builtin_amdgcn_wmma_f32_16x16x32_bf16(false, a_hi, false, b_lo,
                                                             (short)0, acc[t], false, false);
            acc[t] = __builtin_amdgcn_wmma_f32_16x16x32_bf16(false, a_lo, false, b_hi,
                                                             (short)0, acc[t], false, false);
        }
    }

    vq_epilogue(acc, z2, e2, rowmin, colmin, cmin_lds, rowbase, colbase, l16, lg, tid);
}

// -------- fallback path: convert f32 -> bf16 hi/lo in-register (only needs
// 192KB of ws) ---------------------------------------------------------------
__global__ __launch_bounds__(256) void vq_wmma_conv_kernel(const float* __restrict__ z,
                                                           const float* __restrict__ e,
                                                           const float* __restrict__ z2,
                                                           const float* __restrict__ e2,
                                                           unsigned int* __restrict__ rowmin,
                                                           unsigned int* __restrict__ colmin) {
    __shared__ unsigned int cmin_lds[256];
    const int tid = threadIdx.x;
    cmin_lds[tid] = INF_BITS;
    __syncthreads();

    const int lane = tid & 31;
    const int wave = tid >> 5;
    const int l16  = lane & 15;
    const int lg   = lane >> 4;
    const int colbase = blockIdx.x * 256;
    const int rowbase = blockIdx.y * 128 + wave * 16;

    v8f acc[16];
#pragma unroll
    for (int t = 0; t < 16; ++t) acc[t] = (v8f){0.f,0.f,0.f,0.f,0.f,0.f,0.f,0.f};

    const float* zrow  = z + (long)(rowbase + l16) * DIM + lg * 8;
    const float* ecol0 = e + (long)(colbase + l16) * DIM + lg * 8;

#pragma unroll 1
    for (int s = 0; s < DIM / 32; ++s) {
        const int k0 = s * 32;
        v16bf a_hi, a_lo;
        {
            float4 u0 = *(const float4*)(zrow + k0);
            float4 u1 = *(const float4*)(zrow + k0 + 4);
            float4 u2 = *(const float4*)(zrow + k0 + 16);
            float4 u3 = *(const float4*)(zrow + k0 + 20);
            cvt_split8(u0, u1, a_hi, a_lo, 0);
            cvt_split8(u2, u3, a_hi, a_lo, 8);
        }
#pragma unroll
        for (int t = 0; t < 16; ++t) {
            const float* ep = ecol0 + (long)t * 16 * DIM + k0;
            float4 b0 = *(const float4*)(ep);
            float4 b1 = *(const float4*)(ep + 4);
            float4 b2 = *(const float4*)(ep + 16);
            float4 b3 = *(const float4*)(ep + 20);
            v16bf b_hi, b_lo;
            cvt_split8(b0, b1, b_hi, b_lo, 0);
            cvt_split8(b2, b3, b_hi, b_lo, 8);
            acc[t] = __builtin_amdgcn_wmma_f32_16x16x32_bf16(false, a_hi, false, b_hi,
                                                             (short)0, acc[t], false, false);
            acc[t] = __builtin_amdgcn_wmma_f32_16x16x32_bf16(false, a_hi, false, b_lo,
                                                             (short)0, acc[t], false, false);
            acc[t] = __builtin_amdgcn_wmma_f32_16x16x32_bf16(false, a_lo, false, b_hi,
                                                             (short)0, acc[t], false, false);
        }
    }

    vq_epilogue(acc, z2, e2, rowmin, colmin, cmin_lds, rowbase, colbase, l16, lg, tid);
}

__global__ __launch_bounds__(256) void vq_reduce_kernel(const unsigned int* __restrict__ rowmin,
                                                        const unsigned int* __restrict__ colmin,
                                                        float* __restrict__ out) {
    __shared__ float sbuf[256];
    int tid = threadIdx.x;
    float sc = 0.0f, sr = 0.0f;
    for (int i = tid; i < NE; i += 256) sc += __uint_as_float(colmin[i]);
    for (int i = tid; i < NZ; i += 256) sr += __uint_as_float(rowmin[i]);
    sbuf[tid] = sc / (float)NE + sr / (float)NZ;
    __syncthreads();
    for (int s2 = 128; s2 > 0; s2 >>= 1) {
        if (tid < s2) sbuf[tid] += sbuf[tid + s2];
        __syncthreads();
    }
    if (tid == 0) out[0] = sbuf[0];
}

extern "C" void kernel_launch(void* const* d_in, const int* in_sizes, int n_in,
                              void* d_out, int out_size, void* d_ws, size_t ws_size,
                              hipStream_t stream) {
    const float* z = (const float*)d_in[0];
    const float* e = (const float*)d_in[1];
    float* out = (float*)d_out;

    float* z2 = (float*)d_ws;
    float* e2 = z2 + NZ;
    unsigned int* rowmin = (unsigned int*)(e2 + NE);
    unsigned int* colmin = rowmin + NZ;

    const size_t base_bytes  = (size_t)(NZ + NE + NZ + NE) * 4;       // 192 KB
    const size_t split_bytes = (size_t)(NZ + NE) * DIM * 2 * 2;       // 24 MB

    vq_init_kernel<<<(NZ + NE + 255) / 256, 256, 0, stream>>>(rowmin);
    vq_norms_kernel<<<(NZ + NE) / 8, 256, 0, stream>>>(z, e, z2, e2);

    dim3 grid(NE / 256, NZ / 128);
    if (ws_size >= base_bytes + split_bytes) {
        __bf16* zhi = (__bf16*)((char*)d_ws + base_bytes);
        __bf16* zlo = zhi + (long)NZ * DIM;
        __bf16* ehi = zlo + (long)NZ * DIM;
        __bf16* elo = ehi + (long)NE * DIM;
        vq_split_kernel<<<(NZ + NE) * DIM / 8 / 256, 256, 0, stream>>>(z, e, zhi, zlo, ehi, elo);
        vq_wmma_fast_kernel<<<grid, 256, 0, stream>>>(zhi, zlo, ehi, elo, z2, e2, rowmin, colmin);
    } else {
        vq_wmma_conv_kernel<<<grid, 256, 0, stream>>>(z, e, z2, e2, rowmin, colmin);
    }

    vq_reduce_kernel<<<1, 256, 0, stream>>>(rowmin, colmin, out);
}
